// HierarchicalSkeletalEncoder_23613730193795
// MI455X (gfx1250) — matrix operand order repaired and checked by reading, not compiled, not run on hardware
//
#include <hip/hip_runtime.h>
#include <hip/hip_bf16.h>

typedef __attribute__((ext_vector_type(16))) _Float16 v16h;
typedef __attribute__((ext_vector_type(8)))  _Float16 v8h;
typedef __attribute__((ext_vector_type(8)))  float    v8f;

#define NJ    17
#define HD    64
#define TS    16      // samples per workgroup (kernel 1)
#define FSTR  72      // padded feat row stride (halves)
#define ABSTR 136     // padded ab row stride (halves)
#define WSTR  72      // padded WcT row stride (halves)
#define W2STR 392     // padded Wp row stride (halves), kernel 2

// ---- graph tables (precomputed from SUBSETS = [[0,5,6,11,12],[7,8,13,14],[9,10,15,16]]) ----
__constant__ int c_cnt[3][NJ] = {
  {8,0,0,0,0,8,8,5,5,0,0,8,8,5,5,0,0},
  {4,0,0,0,0,4,4,12,12,4,4,4,4,12,12,4,4},
  {0,0,0,0,0,0,0,4,4,7,7,0,0,4,4,7,7}};

__constant__ signed char c_nbr[3][NJ][12] = {
  { // level 0
    {5,6,7,8,11,12,13,14,0,0,0,0},{0,0,0,0,0,0,0,0,0,0,0,0},{0,0,0,0,0,0,0,0,0,0,0,0},
    {0,0,0,0,0,0,0,0,0,0,0,0},{0,0,0,0,0,0,0,0,0,0,0,0},
    {0,6,7,8,11,12,13,14,0,0,0,0},{0,5,7,8,11,12,13,14,0,0,0,0},
    {0,5,6,11,12,0,0,0,0,0,0,0},{0,5,6,11,12,0,0,0,0,0,0,0},
    {0,0,0,0,0,0,0,0,0,0,0,0},{0,0,0,0,0,0,0,0,0,0,0,0},
    {0,5,6,7,8,12,13,14,0,0,0,0},{0,5,6,7,8,11,13,14,0,0,0,0},
    {0,5,6,11,12,0,0,0,0,0,0,0},{0,5,6,11,12,0,0,0,0,0,0,0},
    {0,0,0,0,0,0,0,0,0,0,0,0},{0,0,0,0,0,0,0,0,0,0,0,0}},
  { // level 1
    {7,8,13,14,0,0,0,0,0,0,0,0},{0,0,0,0,0,0,0,0,0,0,0,0},{0,0,0,0,0,0,0,0,0,0,0,0},
    {0,0,0,0,0,0,0,0,0,0,0,0},{0,0,0,0,0,0,0,0,0,0,0,0},
    {7,8,13,14,0,0,0,0,0,0,0,0},{7,8,13,14,0,0,0,0,0,0,0,0},
    {0,5,6,8,9,10,11,12,13,14,15,16},{0,5,6,7,9,10,11,12,13,14,15,16},
    {7,8,13,14,0,0,0,0,0,0,0,0},{7,8,13,14,0,0,0,0,0,0,0,0},
    {7,8,13,14,0,0,0,0,0,0,0,0},{7,8,13,14,0,0,0,0,0,0,0,0},
    {0,5,6,7,8,9,10,11,12,14,15,16},{0,5,6,7,8,9,10,11,12,13,15,16},
    {7,8,13,14,0,0,0,0,0,0,0,0},{7,8,13,14,0,0,0,0,0,0,0,0}},
  { // level 2
    {0,0,0,0,0,0,0,0,0,0,0,0},{0,0,0,0,0,0,0,0,0,0,0,0},{0,0,0,0,0,0,0,0,0,0,0,0},
    {0,0,0,0,0,0,0,0,0,0,0,0},{0,0,0,0,0,0,0,0,0,0,0,0},
    {0,0,0,0,0,0,0,0,0,0,0,0},{0,0,0,0,0,0,0,0,0,0,0,0},
    {9,10,15,16,0,0,0,0,0,0,0,0},{9,10,15,16,0,0,0,0,0,0,0,0},
    {7,8,10,13,14,15,16,0,0,0,0,0},{7,8,9,13,14,15,16,0,0,0,0,0},
    {0,0,0,0,0,0,0,0,0,0,0,0},{0,0,0,0,0,0,0,0,0,0,0,0},
    {9,10,15,16,0,0,0,0,0,0,0,0},{9,10,15,16,0,0,0,0,0,0,0,0},
    {7,8,9,10,13,14,16,0,0,0,0,0},{7,8,9,10,13,14,15,0,0,0,0,0}}};

__constant__ signed char c_sub[3][NJ] = {
  {1,0,0,0,0,1,1,0,0,0,0,1,1,0,0,0,0},
  {0,0,0,0,0,0,0,1,1,0,0,0,0,1,1,0,0},
  {0,0,0,0,0,0,0,0,0,1,1,0,0,0,0,1,1}};

__constant__ int c_subcnt[3] = {5,4,4};

__device__ __forceinline__ v16h pack16(v8h lo, v8h hi) {
  return __builtin_shufflevector(lo, hi, 0,1,2,3,4,5,6,7,8,9,10,11,12,13,14,15);
}

// =====================================================================
// Kernel 1: fused per-sample encoder.  16 samples per workgroup.
// =====================================================================
__global__ __launch_bounds__(256, 1)
void k_encode(const float* __restrict__ kp, const float* __restrict__ sc,
              const float* __restrict__ W0, const float* __restrict__ b0,
              const float* __restrict__ W1, const float* __restrict__ b1,
              const float* __restrict__ W2, const float* __restrict__ b2,
              const float* __restrict__ We, const float* __restrict__ be,
              _Float16* __restrict__ pooled)
{
  extern __shared__ char smem[];
  _Float16* feat = (_Float16*)smem;                 // [NJ][TS][FSTR]
  _Float16* ab   = feat + NJ * TS * FSTR;           // [NJ][TS][ABSTR] (a:0..63, b:64..127)
  _Float16* WcT  = ab   + NJ * TS * ABSTR;          // [128][WSTR]  combined [A;B]^T, k-contig
  float*    Pl   = (float*)(WcT + 128 * WSTR);      // [TS][NJ][4]  (x, y, score)
  float*    mnmx = Pl + TS * NJ * 4;                // [TS][4]

  const int tid = threadIdx.x;
  const int n0  = blockIdx.x * TS;

  // Stage combined edge weight, k-contiguous per output column.
  // col<64 : A[col][k] = We[col][k] - We[col][64+k];  col>=64 : B[col-64][k] = We[col-64][64+k]
  for (int idx = tid; idx < 128 * HD; idx += 256) {
    int col = idx >> 6, k = idx & 63;
    float v = (col < 64) ? (We[col * 128 + k] - We[col * 128 + 64 + k])
                         : We[(col - 64) * 128 + 64 + k];
    WcT[col * WSTR + k] = (_Float16)v;
  }

  // Per-sample per-dim min/max over joints
  if (tid < TS) {
    const float* row = kp + (size_t)(n0 + tid) * NJ * 2;
    float mnx = row[0], mxx = row[0], mny = row[1], mxy = row[1];
    for (int j = 1; j < NJ; ++j) {
      float x = row[2 * j], y = row[2 * j + 1];
      mnx = fminf(mnx, x); mxx = fmaxf(mxx, x);
      mny = fminf(mny, y); mxy = fmaxf(mxy, y);
    }
    mnmx[tid * 4 + 0] = mnx; mnmx[tid * 4 + 1] = 1.0f / (mxx - mnx + 1e-6f);
    mnmx[tid * 4 + 2] = mny; mnmx[tid * 4 + 3] = 1.0f / (mxy - mny + 1e-6f);
  }
  __syncthreads();

  // Normalized P = (kx, ky, score)
  for (int idx = tid; idx < TS * NJ; idx += 256) {
    int s = idx / NJ, j = idx % NJ;
    float x  = kp[(size_t)(n0 + s) * NJ * 2 + 2 * j];
    float y  = kp[(size_t)(n0 + s) * NJ * 2 + 2 * j + 1];
    float sv = sc[(size_t)(n0 + s) * NJ + j];
    Pl[(s * NJ + j) * 4 + 0] = (x - mnmx[s * 4 + 0]) * mnmx[s * 4 + 1];
    Pl[(s * NJ + j) * 4 + 1] = (y - mnmx[s * 4 + 2]) * mnmx[s * 4 + 3];
    Pl[(s * NJ + j) * 4 + 2] = sv;
  }
  __syncthreads();

  const int s   = tid >> 4;   // sample 0..15
  const int cg  = tid & 15;   // channel group
  const int ch  = cg * 4;     // 4 channels per thread
  const int lane  = tid & 31;
  const int wave  = tid >> 5;
  const int mhalf = lane >> 4;
  const int lm    = lane & 15;

  float be4[4];
  for (int c = 0; c < 4; ++c) be4[c] = be[ch + c];

  for (int lvl = 0; lvl < 3; ++lvl) {
    const float* W  = (lvl == 0) ? W0 : ((lvl == 1) ? W1 : W2);
    const float* Bb = (lvl == 0) ? b0 : ((lvl == 1) ? b1 : b2);
    float w[4][3], bb[4];
    for (int c = 0; c < 4; ++c) {
      w[c][0] = W[(ch + c) * 3 + 0];
      w[c][1] = W[(ch + c) * 3 + 1];
      w[c][2] = W[(ch + c) * 3 + 2];
      bb[c]   = Bb[ch + c];
    }

    // m = relu(W P + b) * score ; msum = sum over subset joints
    float msum[4] = {0.f, 0.f, 0.f, 0.f};
    for (int j = 0; j < NJ; ++j) {
      float px = Pl[(s * NJ + j) * 4 + 0];
      float py = Pl[(s * NJ + j) * 4 + 1];
      float ps = Pl[(s * NJ + j) * 4 + 2];
      for (int c = 0; c < 4; ++c) {
        float h = fmaxf(w[c][0] * px + w[c][1] * py + w[c][2] * ps + bb[c], 0.f);
        float m = h * ps;
        if (c_sub[lvl][j]) msum[c] += m;
        feat[(j * TS + s) * FSTR + ch + c] = (_Float16)m;
      }
    }
    // feat = H * score ; clique identity: H[u] = msum - m[u] (u in subset), else 0
    for (int j = 0; j < NJ; ++j) {
      float ps = Pl[(s * NJ + j) * 4 + 2];
      for (int c = 0; c < 4; ++c) {
        float m = (float)feat[(j * TS + s) * FSTR + ch + c];
        float f = c_sub[lvl][j] ? (msum[c] - m) * ps : 0.f;
        feat[(j * TS + s) * FSTR + ch + c] = (_Float16)f;
      }
    }
    __syncthreads();

    // WMMA: ab[j][s][0:128] = feat[j][s][0:64] x WcT   (17 joints x 8 n-tiles, K=64)
    for (int t = wave; t < NJ * 8; t += 8) {
      int j = t >> 3, nt = t & 7;
      v8f acc = {0.f, 0.f, 0.f, 0.f, 0.f, 0.f, 0.f, 0.f};
      for (int ks = 0; ks < 2; ++ks) {
        const _Float16* pa = &feat[(j * TS + lm) * FSTR + ks * 32 + mhalf * 8];
        v8h alo = *(const v8h*)pa;
        v8h ahi = *(const v8h*)(pa + 16);
        const _Float16* pb = &WcT[(nt * 16 + lm) * WSTR + ks * 32 + mhalf * 8];
        v8h blo = *(const v8h*)pb;
        v8h bhi = *(const v8h*)(pb + 16);
        acc = __builtin_amdgcn_wmma_f32_16x16x32_f16(
            false, pack16(alo, ahi), false, pack16(blo, bhi),
            (short)0, acc, false, false);
      }
      for (int r = 0; r < 8; ++r) {
        int ss = r + 8 * mhalf;
        ab[(j * TS + ss) * ABSTR + nt * 16 + lm] = (_Float16)acc[r];
      }
    }
    __syncthreads();

    // Neighbor max-pool: Z[j] = relu(max_k(a[j] + b[nbr_k] + be)), 0 if isolated
    float Zsum[4] = {0.f, 0.f, 0.f, 0.f};
    for (int j = 0; j < NJ; ++j) {
      int cnt = c_cnt[lvl][j];
      if (cnt == 0) continue;
      float a0[4], mx[4];
      for (int c = 0; c < 4; ++c) {
        a0[c] = (float)ab[(j * TS + s) * ABSTR + ch + c];
        mx[c] = -3.0e38f;
      }
      for (int k = 0; k < cnt; ++k) {
        int v = c_nbr[lvl][j][k];
        for (int c = 0; c < 4; ++c) {
          float val = a0[c] + (float)ab[(v * TS + s) * ABSTR + 64 + ch + c] + be4[c];
          mx[c] = fmaxf(mx[c], val);
        }
      }
      for (int c = 0; c < 4; ++c) Zsum[c] += fmaxf(mx[c], 0.f);
    }

    // pooled[lvl] = [mean_j H, mean_j Z];  mean_j H = (|S|-1)*msum/17
    float hsc = (float)(c_subcnt[lvl] - 1) * (1.0f / 17.0f);
    _Float16* prow = pooled + (size_t)(n0 + s) * 384 + lvl * 128;
    for (int c = 0; c < 4; ++c) {
      prow[ch + c]      = (_Float16)(msum[c] * hsc);
      prow[64 + ch + c] = (_Float16)(Zsum[c] * (1.0f / 17.0f));
    }
    __syncthreads();   // feat/ab reused next level
  }
}

// =====================================================================
// Kernel 2: head GEMM  out[N][128] = pooled[N][384] @ Wp^T + bp   (WMMA)
// 128 samples per workgroup, 8 waves x (8 Mtiles x 8 Ntiles) tasks.
// =====================================================================
__global__ __launch_bounds__(256, 1)
void k_head(const _Float16* __restrict__ pooled, const float* __restrict__ Wp,
            const float* __restrict__ bp, float* __restrict__ out)
{
  extern __shared__ char smem[];
  _Float16* Wph = (_Float16*)smem;   // [128][W2STR], row o, k-contiguous

  const int tid = threadIdx.x;
  for (int idx = tid; idx < 128 * 384; idx += 256) {
    int o = idx / 384, k = idx % 384;
    Wph[o * W2STR + k] = (_Float16)Wp[o * 384 + k];
  }
  __syncthreads();

  const int lane  = tid & 31;
  const int wave  = tid >> 5;
  const int mhalf = lane >> 4;
  const int lm    = lane & 15;
  const int n0    = blockIdx.x * 128;

  for (int t = wave; t < 64; t += 8) {
    int mt = t >> 3, nt = t & 7;
    int m0 = n0 + mt * 16;
    v8f acc = {0.f, 0.f, 0.f, 0.f, 0.f, 0.f, 0.f, 0.f};
    for (int ks = 0; ks < 12; ++ks) {
      const _Float16* pa = pooled + (size_t)(m0 + lm) * 384 + ks * 32 + mhalf * 8;
      v8h alo = *(const v8h*)pa;
      v8h ahi = *(const v8h*)(pa + 16);
      const _Float16* pb = &Wph[(nt * 16 + lm) * W2STR + ks * 32 + mhalf * 8];
      v8h blo = *(const v8h*)pb;
      v8h bhi = *(const v8h*)(pb + 16);
      acc = __builtin_amdgcn_wmma_f32_16x16x32_f16(
          false, pack16(alo, ahi), false, pack16(blo, bhi),
          (short)0, acc, false, false);
    }
    float bias = bp[nt * 16 + lm];
    for (int r = 0; r < 8; ++r) {
      int row = m0 + r + 8 * mhalf;
      out[(size_t)row * 128 + nt * 16 + lm] = acc[r] + bias;
    }
  }
}

// =====================================================================
extern "C" void kernel_launch(void* const* d_in, const int* in_sizes, int n_in,
                              void* d_out, int out_size, void* d_ws, size_t ws_size,
                              hipStream_t stream) {
  const float* kp = (const float*)d_in[0];
  const float* sc = (const float*)d_in[1];
  const float* W0 = (const float*)d_in[2];
  const float* b0 = (const float*)d_in[3];
  const float* W1 = (const float*)d_in[4];
  const float* b1 = (const float*)d_in[5];
  const float* W2 = (const float*)d_in[6];
  const float* b2 = (const float*)d_in[7];
  const float* We = (const float*)d_in[8];
  const float* be = (const float*)d_in[9];
  const float* Wp = (const float*)d_in[10];
  const float* bp = (const float*)d_in[11];
  float* out = (float*)d_out;

  const int N = in_sizes[1] / NJ;              // 8192
  _Float16* pooled = (_Float16*)d_ws;          // [N][384] f16

  size_t lds1 = (size_t)(NJ * TS * FSTR + NJ * TS * ABSTR + 128 * WSTR) * sizeof(_Float16)
              + (size_t)(TS * NJ * 4 + TS * 4) * sizeof(float);
  size_t lds2 = (size_t)(128 * W2STR) * sizeof(_Float16);

  k_encode<<<N / TS, 256, lds1, stream>>>(kp, sc, W0, b0, W1, b1, W2, b2, We, be, pooled);
  k_head<<<N / 128, 256, lds2, stream>>>(pooled, Wp, bp, out);
}
